// InputAttention_73495480369368
// MI455X (gfx1250) — matrix operand order repaired and checked by reading, not compiled, not run on hardware
//
#include <hip/hip_runtime.h>
#include <hip/hip_bf16.h>

// Problem constants (from reference): S=256, B=8, H=512, K=128, D=512
#define S_LEN 256
#define B_SZ  8
#define H_DIM 512
#define K_LEN 128
#define D_DIM 512

typedef __attribute__((ext_vector_type(16))) __bf16 v16bf;
typedef __attribute__((ext_vector_type(8)))  __bf16 v8bf;
typedef __attribute__((ext_vector_type(8)))  float  v8f;

__device__ __forceinline__ v16bf cat8(v8bf lo, v8bf hi) {
  return __builtin_shufflevector(lo, hi, 0,1,2,3,4,5,6,7,8,9,10,11,12,13,14,15);
}

// CDNA5 hardware tanh (TRANS pipe). Trailing v_nop covers the 1-op TRANS
// result hazard the compiler cannot see through inline asm.
__device__ __forceinline__ float htanh(float x) {
  float r;
  asm("v_tanh_f32 %0, %1\n\tv_nop" : "=v"(r) : "v"(x));
  return r;
}

// CDNA5 async global->LDS copy (16B per lane), tracked on ASYNCcnt.
__device__ __forceinline__ void async_copy16(unsigned int lds_addr, const void* gaddr) {
  asm volatile("global_load_async_to_lds_b128 %0, %1, off"
               :: "v"(lds_addr), "v"(gaddr) : "memory");
}
__device__ __forceinline__ void wait_async0() {
  asm volatile("s_wait_asynccnt 0x0" ::: "memory");
}
__device__ __forceinline__ unsigned int lds_lo32(const void* p) {
  // generic LDS address: low 32 bits are the LDS byte offset (ISA aperture rule)
  return (unsigned int)(unsigned long long)p;
}

// ---------------------------------------------------------------------------
// Generic WMMA GEMM:  C[m,n] = sum_k A[m,k] * Bt[n,k]  (+ bias[n])
// A: bf16 row-major (lda), Bt: bf16 N-major/K-contiguous (ldb), C: f32 (ldc).
// One wave -> 16(m) x 64(n) output strip = 4 WMMA tiles sharing the A frag.
// blockDim = 256 (8 waves). Optional batching via blockIdx.z + strides.
// ---------------------------------------------------------------------------
__global__ void __launch_bounds__(256)
gemm_bt_bf16(const __bf16* __restrict__ A, const __bf16* __restrict__ Bt,
             const float* __restrict__ bias, float* __restrict__ C,
             int M, int N, int Kd, int lda, int ldb, int ldc,
             long sAb, long sBb, long sCb)
{
  const int lane = threadIdx.x & 31;
  const int wave = threadIdx.x >> 5;
  const int tilesN = N >> 6;                    // 64-wide wave strips
  const int wt = blockIdx.x * 8 + wave;
  const int mt = wt / tilesN;
  const int nt = wt % tilesN;
  if (mt * 16 >= M) return;
  A  += (long)blockIdx.z * sAb;
  Bt += (long)blockIdx.z * sBb;
  C  += (long)blockIdx.z * sCb;
  const int m0 = mt * 16, n0 = nt * 64;
  const int r16  = lane & 15;
  const int half = lane >> 4;

  // 16-bit A frag (16x32): lane holds row m0+r16; K = half*16 + 0..15 contiguous
  const __bf16* arow = A + (size_t)(m0 + r16) * lda + half * 16;
  // 16-bit B frag (32x16): lane holds col n+r16; K = half*16 + 0..15 contiguous
  const __bf16* b0 = Bt + (size_t)(n0 +  0 + r16) * ldb + half * 16;
  const __bf16* b1 = Bt + (size_t)(n0 + 16 + r16) * ldb + half * 16;
  const __bf16* b2 = Bt + (size_t)(n0 + 32 + r16) * ldb + half * 16;
  const __bf16* b3 = Bt + (size_t)(n0 + 48 + r16) * ldb + half * 16;

  v8f acc0 = {}, acc1 = {}, acc2 = {}, acc3 = {};
  for (int k0 = 0; k0 < Kd; k0 += 32) {
    v16bf af  = cat8(*(const v8bf*)(arow + k0), *(const v8bf*)(arow + k0 + 8));
    v16bf bf0 = cat8(*(const v8bf*)(b0 + k0),   *(const v8bf*)(b0 + k0 + 8));
    v16bf bf1 = cat8(*(const v8bf*)(b1 + k0),   *(const v8bf*)(b1 + k0 + 8));
    v16bf bf2 = cat8(*(const v8bf*)(b2 + k0),   *(const v8bf*)(b2 + k0 + 8));
    v16bf bf3 = cat8(*(const v8bf*)(b3 + k0),   *(const v8bf*)(b3 + k0 + 8));
    acc0 = __builtin_amdgcn_wmma_f32_16x16x32_bf16(false, af, false, bf0, (short)0, acc0, false, false);
    acc1 = __builtin_amdgcn_wmma_f32_16x16x32_bf16(false, af, false, bf1, (short)0, acc1, false, false);
    acc2 = __builtin_amdgcn_wmma_f32_16x16x32_bf16(false, af, false, bf2, (short)0, acc2, false, false);
    acc3 = __builtin_amdgcn_wmma_f32_16x16x32_bf16(false, af, false, bf3, (short)0, acc3, false, false);
  }

  // C/D layout: VGPR r -> row m0 + (lane<16?0:8) + r, col n0 + lane%16
  const int rbase = m0 + half * 8;
  v8f accs[4] = {acc0, acc1, acc2, acc3};
#pragma unroll
  for (int t = 0; t < 4; ++t) {
    const int col = n0 + t * 16 + r16;
    const float bv = bias ? bias[col] : 0.f;
#pragma unroll
    for (int r = 0; r < 8; ++r)
      C[(size_t)(rbase + r) * ldc + col] = accs[t][r] + bv;
  }
}

// ---------------------------------------------------------------------------
// Prep kernels: fp32 -> bf16 conversions and the embedding gather.
// ---------------------------------------------------------------------------
__global__ void __launch_bounds__(256)
f32_to_bf16_kernel(const float* __restrict__ in, __bf16* __restrict__ out, int n) {
  int i = blockIdx.x * 256 + threadIdx.x;
  if (i < n) out[i] = (__bf16)in[i];
}

// memory_bank (S,B,H) f32  ->  h_t_bf16 rows (b*S+s) x H
__global__ void __launch_bounds__(256)
convert_ht_kernel(const float* __restrict__ mb, __bf16* __restrict__ ht) {
  const int s = blockIdx.x / B_SZ, b = blockIdx.x % B_SZ;
  const float* src = mb + ((size_t)s * B_SZ + b) * H_DIM;
  __bf16* dst = ht + ((size_t)b * S_LEN + s) * H_DIM;
  for (int h = threadIdx.x; h < H_DIM; h += 256) dst[h] = (__bf16)src[h];
}

// know (K,B,1) int -> hs_bf16 rows (b*K+k) x D  and  hsT_bf16[b][d][k] (for k_star GEMM)
__global__ void __launch_bounds__(256)
gather_hs_kernel(const int* __restrict__ know, const float* __restrict__ emb,
                 __bf16* __restrict__ hs, __bf16* __restrict__ hsT) {
  const int b = blockIdx.x / K_LEN, k = blockIdx.x % K_LEN;
  const int tok = know[(size_t)k * B_SZ + b];
  const float* e = emb + (size_t)tok * D_DIM;
  __bf16* dst  = hs  + ((size_t)b * K_LEN + k) * D_DIM;
  __bf16* dstT = hsT + (size_t)b * D_DIM * K_LEN + k;
  for (int d = threadIdx.x; d < D_DIM; d += 256) {
    __bf16 x = (__bf16)e[d];
    dst[d] = x;
    dstT[(size_t)d * K_LEN] = x;
  }
}

// ---------------------------------------------------------------------------
// Fused score + mask + softmax, s-tiled with async double-buffered LDS
// staging of uh[b].
//   block = (s_tile of 8, b); 8 waves, wave w owns s = s_tile*8 + w.
//   uh[b] (128 x 512 f32) streamed through LDS in 16 chunks of 8 rows via
//   global_load_async_to_lds_b128 (ASYNCcnt), overlapping the v_tanh_f32
//   compute of chunk c with the copy of chunk c+1.
//   score[k] = sum_h v[h] * tanh(wq[b,s,h] + uh[b,k,h]); per-wave masked
//   softmax over K=128 with wave32 shuffles.
// ---------------------------------------------------------------------------
#define CROWS  8                       // k-rows per chunk
#define NCHUNK (K_LEN / CROWS)         // 16 chunks

__global__ void __launch_bounds__(256)
score_softmax_kernel(const float* __restrict__ wq, const float* __restrict__ uh,
                     const float* __restrict__ v, const int* __restrict__ know_lengths,
                     float* __restrict__ align_out, __bf16* __restrict__ align_bf)
{
  __shared__ float uh_buf[2][CROWS * H_DIM];   // 2 x 16 KB
  __shared__ float s_sc[8][K_LEN];             // 4 KB

  const int b    = blockIdx.y;
  const int tid  = threadIdx.x;
  const int lane = tid & 31, wave = tid >> 5;
  const int hbase = lane * 16;
  const int row  = b * S_LEN + blockIdx.x * 8 + wave;   // b*S + s

  float wqv[16], vv[16];
  const float* wqp = wq + (size_t)row * H_DIM + hbase;
#pragma unroll
  for (int i = 0; i < 16; ++i) { wqv[i] = wqp[i]; vv[i] = v[hbase + i]; }

  const char* ubase = (const char*)(uh + (size_t)b * K_LEN * H_DIM);

  // issue chunk 0 (4 x 16B per thread = 16 KB per chunk)
  {
    const unsigned int l0 = lds_lo32(&uh_buf[0][0]);
#pragma unroll
    for (int j = 0; j < 4; ++j) {
      const int u = tid + 256 * j;
      async_copy16(l0 + u * 16, ubase + (size_t)u * 16);
    }
  }

  for (int c = 0; c < NCHUNK; ++c) {
    const int cur = c & 1;
    wait_async0();          // own chunk-c copies done (in-order on ASYNCcnt)
    __syncthreads();        // everyone's copies visible; prev compute finished
    if (c + 1 < NCHUNK) {   // overlap: start chunk c+1 into the other buffer
      const unsigned int ln = lds_lo32(&uh_buf[cur ^ 1][0]);
      const char* gsrc = ubase + (size_t)(c + 1) * CROWS * H_DIM * 4;
#pragma unroll
      for (int j = 0; j < 4; ++j) {
        const int u = tid + 256 * j;
        async_copy16(ln + u * 16, gsrc + (size_t)u * 16);
      }
    }
    // compute the 8 k-rows of this chunk for this wave's s
#pragma unroll
    for (int kk = 0; kk < CROWS; ++kk) {
      const float* up = &uh_buf[cur][kk * H_DIM + hbase];
      float acc = 0.f;
#pragma unroll
      for (int i = 0; i < 16; ++i)
        acc += vv[i] * htanh(wqv[i] + up[i]);
#pragma unroll
      for (int off = 16; off > 0; off >>= 1)   // wave32 reduction
        acc += __shfl_xor(acc, off, 32);
      if (lane == 0) s_sc[wave][c * CROWS + kk] = acc;
    }
  }

  // per-wave masked softmax over K=128 (lane owns k = lane + 32*j); purely
  // intra-wave: s_sc[wave] was written by this wave only.
  const int klen = know_lengths[b];
  float sc[4];
#pragma unroll
  for (int j = 0; j < 4; ++j) {
    const int k = lane + 32 * j;
    sc[j] = (k < klen) ? s_sc[wave][k] : -3.402823466e38f;
  }
  float m = fmaxf(fmaxf(sc[0], sc[1]), fmaxf(sc[2], sc[3]));
#pragma unroll
  for (int off = 16; off > 0; off >>= 1)
    m = fmaxf(m, __shfl_xor(m, off, 32));
  float e[4], sum = 0.f;
#pragma unroll
  for (int j = 0; j < 4; ++j) {
    const int k = lane + 32 * j;
    e[j] = (k < klen) ? __expf(sc[j] - m) : 0.f;
    sum += e[j];
  }
#pragma unroll
  for (int off = 16; off > 0; off >>= 1)
    sum += __shfl_xor(sum, off, 32);
  const float inv = 1.f / sum;
#pragma unroll
  for (int j = 0; j < 4; ++j) {
    const int k = lane + 32 * j;
    const float a = e[j] * inv;
    align_out[(size_t)row * K_LEN + k] = a;
    align_bf[(size_t)row * K_LEN + k]  = (__bf16)a;
  }
}

// ---------------------------------------------------------------------------
extern "C" void kernel_launch(void* const* d_in, const int* in_sizes, int n_in,
                              void* d_out, int out_size, void* d_ws, size_t ws_size,
                              hipStream_t stream) {
  (void)in_sizes; (void)n_in; (void)out_size; (void)ws_size;
  const float* memory_bank = (const float*)d_in[0];
  // d_in[1] = src_lengths (unused by the reference output)
  const int*   know        = (const int*)d_in[2];
  const int*   know_len    = (const int*)d_in[3];
  const float* emb         = (const float*)d_in[4];
  const float* W_h         = (const float*)d_in[5];
  const float* b_h         = (const float*)d_in[6];
  const float* W_k         = (const float*)d_in[7];
  const float* v           = (const float*)d_in[8];

  // workspace carve-up (256B aligned)
  char* ws = (char*)d_ws;
  size_t off = 0;
  auto carve = [&](size_t bytes) -> void* {
    void* p = ws + off;
    off += (bytes + 255) & ~(size_t)255;
    return p;
  };
  __bf16* ht_bf   = (__bf16*)carve((size_t)B_SZ * S_LEN * H_DIM * 2); // 2 MB
  __bf16* hs_bf   = (__bf16*)carve((size_t)B_SZ * K_LEN * D_DIM * 2); // 1 MB
  __bf16* hsT_bf  = (__bf16*)carve((size_t)B_SZ * D_DIM * K_LEN * 2); // 1 MB
  __bf16* Wh_bf   = (__bf16*)carve((size_t)H_DIM * H_DIM * 2);        // 0.5 MB
  __bf16* Wk_bf   = (__bf16*)carve((size_t)H_DIM * D_DIM * 2);        // 0.5 MB
  float*  wq      = (float*) carve((size_t)B_SZ * S_LEN * H_DIM * 4); // 4 MB
  float*  uh      = (float*) carve((size_t)B_SZ * K_LEN * H_DIM * 4); // 2 MB
  __bf16* algn_bf = (__bf16*)carve((size_t)B_SZ * S_LEN * K_LEN * 2); // 0.5 MB

  float* kstar     = (float*)d_out;                                   // (B,S,D)
  float* align_out = (float*)d_out + (size_t)B_SZ * S_LEN * D_DIM;    // (B,S,K)

  // 1) conversions + gather
  f32_to_bf16_kernel<<<(H_DIM * H_DIM + 255) / 256, 256, 0, stream>>>(W_h, Wh_bf, H_DIM * H_DIM);
  f32_to_bf16_kernel<<<(H_DIM * D_DIM + 255) / 256, 256, 0, stream>>>(W_k, Wk_bf, H_DIM * D_DIM);
  convert_ht_kernel<<<S_LEN * B_SZ, 256, 0, stream>>>(memory_bank, ht_bf);
  gather_hs_kernel<<<B_SZ * K_LEN, 256, 0, stream>>>(know, emb, hs_bf, hsT_bf);

  // 2) wq = h_t @ W_h^T + b_h   (M=2048, N=512, K=512) -> 1024 wave-strips / 8 = 128 blocks
  gemm_bt_bf16<<<dim3(128, 1, 1), 256, 0, stream>>>(
      ht_bf, Wh_bf, b_h, wq, B_SZ * S_LEN, H_DIM, H_DIM, H_DIM, H_DIM, H_DIM, 0, 0, 0);
  // 3) uh = h_s @ W_k^T          (M=1024, N=512, K=512) -> 64 blocks
  gemm_bt_bf16<<<dim3(64, 1, 1), 256, 0, stream>>>(
      hs_bf, Wk_bf, nullptr, uh, B_SZ * K_LEN, H_DIM, D_DIM, D_DIM, D_DIM, H_DIM, 0, 0, 0);

  // 4) fused additive score + masked softmax (async LDS staging + v_tanh_f32)
  score_softmax_kernel<<<dim3(S_LEN / 8, B_SZ), 256, 0, stream>>>(
      wq, uh, v, know_len, align_out, algn_bf);

  // 5) k_star[b] = align[b] @ h_s[b]   (batched M=256, N=512, K=128) -> 16 blocks x 8 batches
  gemm_bt_bf16<<<dim3(16, 1, B_SZ), 256, 0, stream>>>(
      algn_bf, hsT_bf, nullptr, kstar, S_LEN, D_DIM, K_LEN, K_LEN, K_LEN, D_DIM,
      (long)S_LEN * K_LEN, (long)D_DIM * K_LEN, (long)S_LEN * D_DIM);
}
